// CX_Model_32332513804392
// MI455X (gfx1250) — matrix operand order repaired and checked by reading, not compiled, not run on hardware
//
#include <hip/hip_runtime.h>

// CDNA5 (gfx1250) wave32 WMMA types
typedef __attribute__((ext_vector_type(16))) _Float16 v16h;
typedef __attribute__((ext_vector_type(8)))  _Float16 v8h;
typedef __attribute__((ext_vector_type(8)))  float    v8f;

// ---------------------------------------------------------------------------
// Kernel P: pre-arrange the fixed 256x16 matrix M[(i*16+k),o] = We2[k, i*16+o]
// (plus a 9th K=32 block carrying the be2 bias matrix) into per-lane WMMA
// B-fragment order: bpack[t*512 + lane*16 + e] = M[32t + 16(e>>3) + 8g + (e&7)][n],
// n = lane&15, g = lane>>4.  9 * 32 lanes * 16 halves = 4608 elements.
// ---------------------------------------------------------------------------
__global__ void pack_B_kernel(const float* __restrict__ We2,
                              const float* __restrict__ be2,
                              _Float16* __restrict__ bpack)
{
    int idx = blockIdx.x * blockDim.x + threadIdx.x;
    if (idx >= 9 * 32 * 16) return;
    int t   = idx >> 9;      // K-block 0..8
    int rem = idx & 511;
    int L   = rem >> 4;      // lane 0..31
    int e   = rem & 15;      // element within v16h
    int n   = L & 15;        // output column
    int g   = L >> 4;        // lane group
    int kl  = (e & 8) * 2 + 8 * g + (e & 7);    // local K within the 32-wide block (B layout)
    float v;
    if (t < 8) {
        int q  = 32 * t + kl;   // global K in 0..255
        int i  = q >> 4;        // h channel
        int ke = q & 15;        // e_hidden channel
        v = We2[ke * 256 + i * 16 + n];
    } else {
        // bias block: A carries h in K=0..15, so B row kl holds be2[kl*16+n]
        v = (kl < 16) ? be2[kl * 16 + n] : 0.0f;
    }
    bpack[idx] = (_Float16)v;
}

// ---------------------------------------------------------------------------
// Kernel A: h = relu(x @ W0 + b0).  One V_WMMA_F32_16X16X32_F16 per 16-node
// tile (K = NODE_F = 32 exactly).  N = 100000 = 6250 * 16 tiles.
// ---------------------------------------------------------------------------
__global__ void node_lin0_kernel(const float* __restrict__ x,
                                 const float* __restrict__ W0,
                                 const float* __restrict__ b0,
                                 float* __restrict__ h, int nTiles)
{
    const int lane = threadIdx.x & 31;
    const int wave = threadIdx.x >> 5;
    const int wpb  = blockDim.x >> 5;
    const int m    = lane & 15;   // row within tile / output column
    const int g    = lane >> 4;   // lane group

    // B fragment for W0 (32x16): element e <-> K = 16*(e>=8) + 8g + (e&7), column n = m
    v16h bw;
    #pragma unroll
    for (int e = 0; e < 16; ++e)
        bw[e] = (_Float16)W0[((e & 8) * 2 + 8 * g + (e & 7)) * 16 + m];
    const float bias = b0[m];

    for (int tile = blockIdx.x * wpb + wave; tile < nTiles;
         tile += gridDim.x * wpb) {
        const float* xr = x + (size_t)(tile * 16 + m) * 32;
        v16h a;
        #pragma unroll
        for (int e = 0; e < 8; ++e) a[e]     = (_Float16)xr[8 * g + e];
        #pragma unroll
        for (int e = 0; e < 8; ++e) a[e + 8] = (_Float16)xr[16 + 8 * g + e];

        v8f c = {};
        c = __builtin_amdgcn_wmma_f32_16x16x32_f16(
                false, a, false, bw, (short)0, c, false, false);

        #pragma unroll
        for (int r = 0; r < 8; ++r) {
            float v = c[r] + bias;
            v = v > 0.0f ? v : 0.0f;
            h[(size_t)(tile * 16 + r + 8 * g) * 16 + m] = v;
        }
    }
}

// ---------------------------------------------------------------------------
// Kernel B: e_hidden = relu(edge_attr @ We1 + be1), stored f16 (two 16B
// vector stores per edge).  K=8 only — 205 MFLOP, memory-shaped; plain VALU.
// ---------------------------------------------------------------------------
__global__ void edge_mlp1_kernel(const float* __restrict__ ea,
                                 const float* __restrict__ We1,
                                 const float* __restrict__ be1,
                                 _Float16* __restrict__ ehid, int E)
{
    int e = blockIdx.x * blockDim.x + threadIdx.x;
    if (e >= E) return;
    float a[8];
    #pragma unroll
    for (int j = 0; j < 8; j += 4) {
        float4 t4 = *(const float4*)(ea + (size_t)e * 8 + j);
        a[j] = t4.x; a[j + 1] = t4.y; a[j + 2] = t4.z; a[j + 3] = t4.w;
    }
    v8h lo, hi;
    #pragma unroll
    for (int j = 0; j < 16; ++j) {
        float s = be1[j];
        #pragma unroll
        for (int k = 0; k < 8; ++k) s += a[k] * We1[k * 16 + j];
        _Float16 r = (_Float16)(s > 0.0f ? s : 0.0f);
        if (j < 8) lo[j] = r; else hi[j - 8] = r;
    }
    *(v8h*)(ehid + (size_t)e * 16)     = lo;
    *(v8h*)(ehid + (size_t)e * 16 + 8) = hi;
}

// ---------------------------------------------------------------------------
// Kernel D: out = h @ Wroot + bconv  (51 MFLOP; also initializes the
// accumulation target for the message scatter every call) and emits the f16
// copy of h (coalesced 32B/row) used by the gather in the message kernel.
// ---------------------------------------------------------------------------
__global__ void node_root_kernel(const float* __restrict__ h,
                                 const float* __restrict__ Wroot,
                                 const float* __restrict__ bconv,
                                 float* __restrict__ out,
                                 _Float16* __restrict__ h16, int N)
{
    int v = blockIdx.x * blockDim.x + threadIdx.x;
    if (v >= N) return;
    float hr[16];
    #pragma unroll
    for (int j = 0; j < 16; j += 4) {
        float4 t4 = *(const float4*)(h + (size_t)v * 16 + j);
        hr[j] = t4.x; hr[j + 1] = t4.y; hr[j + 2] = t4.z; hr[j + 3] = t4.w;
    }
    v8h lo, hi;
    #pragma unroll
    for (int j = 0; j < 8; ++j)  lo[j] = (_Float16)hr[j];
    #pragma unroll
    for (int j = 0; j < 8; ++j)  hi[j] = (_Float16)hr[j + 8];
    *(v8h*)(h16 + (size_t)v * 16)     = lo;
    *(v8h*)(h16 + (size_t)v * 16 + 8) = hi;

    #pragma unroll
    for (int o = 0; o < 16; ++o) {
        float s = bconv[o];
        #pragma unroll
        for (int n = 0; n < 16; ++n) s += hr[n] * Wroot[n * 16 + o];
        out[(size_t)v * 16 + o] = s;
    }
}

// ---------------------------------------------------------------------------
// Kernel C (dominant, 6.55 GFLOP): per 16-edge tile,
//   A[m, i*16+k] = h[src[m]][i] * ehid[m][k]   (outer product, built in regs)
//   msg = A(16x256) @ M(256x16)  -> 8 chained WMMAs + 1 bias-block WMMA,
// then scatter-add msg rows into out[dst] with global f32 atomics.
//
// Gather dedup: lanes L and L+16 share edge row m=L&15, so each lane loads
// only its 16B half of the f16 h[src] row, stages it in a 512B-per-wave LDS
// tile, and reads the full 32B row back (same-wave DS ordering, no barrier).
// All fragment register indices are compile-time to avoid scratch spills.
// ---------------------------------------------------------------------------
__global__ void nnconv_msg_kernel(const int* __restrict__ ei,
                                  const _Float16* __restrict__ h16,
                                  const _Float16* __restrict__ ehid,
                                  const _Float16* __restrict__ bpack,
                                  float* __restrict__ out, int E)
{
    const int lane = threadIdx.x & 31;
    const int wave = threadIdx.x >> 5;
    const int wpb  = blockDim.x >> 5;
    const int m    = lane & 15;
    const int g    = lane >> 4;
    const int nTiles = E >> 4;

    __shared__ __align__(32) _Float16 lds_h[8][16][16];  // [wave][row][chan]

    // Fixed B fragments, loaded once and pinned in VGPRs for the whole loop.
    v16h B[9];
    #pragma unroll
    for (int t = 0; t < 9; ++t)
        B[t] = *(const v16h*)(bpack + ((size_t)t * 32 + lane) * 16);

    for (int tile = blockIdx.x * wpb + wave; tile < nTiles;
         tile += gridDim.x * wpb) {
        const int e0  = tile << 4;
        const int src = ei[e0 + m];           // shared by lane pair
        const int dm  = ei[E + e0 + m];       // dst of "my" edge row

        __builtin_prefetch(ei + e0 + 16, 0, 1);   // global_prefetch_b8

        // each lane fetches its disjoint 16B half of the f16 source row,
        // stages via LDS, reads back the full 32B row
        v8h hhalf = *(const v8h*)(h16 + (size_t)src * 16 + 8 * g);
        *(v8h*)(&lds_h[wave][m][8 * g]) = hhalf;
        v16h hh = *(const v16h*)(&lds_h[wave][m][0]);

        // this lane only ever needs ehid channels 8g..8g+7 of its edge row
        v8h er = *(const v8h*)(ehid + ((size_t)e0 + m) * 16 + 8 * g);

        v8f c = {};
        #pragma unroll
        for (int t = 0; t < 8; ++t) {
            // A element e: K = 32t + (8g + (e&7)) + 16*(e>=8)
            //  -> h channel i = 2t + (e>>3) (compile-time), ehid ch = er[e&7]
            v16h a;
            #pragma unroll
            for (int e = 0; e < 8; ++e) a[e]     = hh[2 * t]     * er[e];
            #pragma unroll
            for (int e = 0; e < 8; ++e) a[e + 8] = hh[2 * t + 1] * er[e];
            c = __builtin_amdgcn_wmma_f32_16x16x32_f16(
                    false, a, false, B[t], (short)0, c, false, false);
        }
        {   // bias block: A = [h_src | 0], B = [be2-matrix ; 0]
            v16h a;
            #pragma unroll
            for (int e = 0; e < 8; ++e) a[e] = g ? hh[e + 8] : hh[e];
            #pragma unroll
            for (int e = 8; e < 16; ++e) a[e] = (_Float16)0.0f;
            c = __builtin_amdgcn_wmma_f32_16x16x32_f16(
                    false, a, false, B[8], (short)0, c, false, false);
        }

        // D row r+8g = edge e0+r+8g, column n = m; fetch that edge's dst
        // from the lane that already loaded it, then scatter-add.
        #pragma unroll
        for (int r = 0; r < 8; ++r) {
            int d = __shfl(dm, r + 8 * g, 32);
            atomicAdd(out + (size_t)d * 16 + m, c[r]);
        }
    }
}

// ---------------------------------------------------------------------------
// Kernel E: edge score head.
//   emb = out[src]*out[dst]; relu(emb@W1+b1) @ W2 + b2
// ---------------------------------------------------------------------------
__global__ void edge_score_kernel(const int* __restrict__ ei,
                                  const float* __restrict__ out,
                                  const float* __restrict__ W1,
                                  const float* __restrict__ b1,
                                  const float* __restrict__ W2,
                                  const float* __restrict__ b2,
                                  float* __restrict__ score, int E)
{
    int e = blockIdx.x * blockDim.x + threadIdx.x;
    if (e >= E) return;
    int s = ei[e], d = ei[E + e];
    float emb[16];
    #pragma unroll
    for (int j = 0; j < 16; j += 4) {
        float4 a4 = *(const float4*)(out + (size_t)s * 16 + j);
        float4 b4 = *(const float4*)(out + (size_t)d * 16 + j);
        emb[j] = a4.x * b4.x; emb[j + 1] = a4.y * b4.y;
        emb[j + 2] = a4.z * b4.z; emb[j + 3] = a4.w * b4.w;
    }
    float acc = b2[0];
    #pragma unroll
    for (int c8 = 0; c8 < 8; ++c8) {
        float t = b1[c8];
        #pragma unroll
        for (int j = 0; j < 16; ++j) t += emb[j] * W1[j * 8 + c8];
        t = t > 0.0f ? t : 0.0f;
        acc += t * W2[c8];
    }
    score[e] = acc;
}

// ---------------------------------------------------------------------------
extern "C" void kernel_launch(void* const* d_in, const int* in_sizes, int n_in,
                              void* d_out, int out_size, void* d_ws, size_t ws_size,
                              hipStream_t stream)
{
    const float* x     = (const float*)d_in[0];
    const int*   ei    = (const int*)  d_in[1];
    const float* ea    = (const float*)d_in[2];
    const float* W0    = (const float*)d_in[3];
    const float* b0    = (const float*)d_in[4];
    const float* We1   = (const float*)d_in[5];
    const float* be1   = (const float*)d_in[6];
    const float* We2   = (const float*)d_in[7];
    const float* be2   = (const float*)d_in[8];
    const float* Wroot = (const float*)d_in[9];
    const float* bconv = (const float*)d_in[10];
    const float* W1    = (const float*)d_in[11];
    const float* b1    = (const float*)d_in[12];
    const float* W2    = (const float*)d_in[13];
    const float* b2    = (const float*)d_in[14];
    float* score = (float*)d_out;

    const int N = in_sizes[0] / 32;   // 100000 (multiple of 16)
    const int E = in_sizes[1] / 2;    // 800000 (multiple of 16)

    // workspace carve-out (256B-aligned): h (6.4MB) | out (6.4MB) |
    // h16 (3.2MB) | ehid f16 (25.6MB) | bpack (9KB)  -> ~41.6MB total
    char*  ws  = (char*)d_ws;
    size_t off = 0;
    auto carve = [&](size_t bytes) {
        char* p = ws + off;
        off = (off + bytes + 255) & ~(size_t)255;
        return p;
    };
    float*    h_buf   = (float*)   carve((size_t)N * 16 * sizeof(float));
    float*    out_buf = (float*)   carve((size_t)N * 16 * sizeof(float));
    _Float16* h16     = (_Float16*)carve((size_t)N * 16 * sizeof(_Float16));
    _Float16* ehid    = (_Float16*)carve((size_t)E * 16 * sizeof(_Float16));
    _Float16* bpack   = (_Float16*)carve((size_t)9 * 512 * sizeof(_Float16));
    (void)ws_size; (void)n_in; (void)out_size;

    const int nodeTiles = N / 16;     // 6250
    const int edgeTiles = E / 16;     // 50000

    pack_B_kernel   <<<(9 * 512 + 255) / 256, 256, 0, stream>>>(We2, be2, bpack);
    node_lin0_kernel<<<(nodeTiles + 7) / 8, 256, 0, stream>>>(x, W0, b0, h_buf, nodeTiles);
    edge_mlp1_kernel<<<(E + 255) / 256, 256, 0, stream>>>(ea, We1, be1, ehid, E);
    node_root_kernel<<<(N + 255) / 256, 256, 0, stream>>>(h_buf, Wroot, bconv, out_buf, h16, N);
    nnconv_msg_kernel<<<(edgeTiles + 7) / 8, 256, 0, stream>>>(ei, h16, ehid, bpack, out_buf, E);
    edge_score_kernel<<<(E + 255) / 256, 256, 0, stream>>>(ei, out_buf, W1, b1, W2, b2, score, E);
}
